// CrossAttention_12120397709751
// MI455X (gfx1250) — compile-verified, matrix operand order checked by reading, not run on hardware
//
#include <hip/hip_runtime.h>
#include <hip/hip_bf16.h>

typedef __attribute__((ext_vector_type(16))) _Float16 v16h;
typedef __attribute__((ext_vector_type(8)))  float    v8f;
typedef __attribute__((ext_vector_type(4)))  unsigned int v4u;
typedef __attribute__((ext_vector_type(4)))  int v4i;

union Frag16 { v16h h; v4u u[2]; };
union Pack8h { _Float16 t[8]; v4u u; };

#define WMMA_F16(A, B, C) \
  __builtin_amdgcn_wmma_f32_16x16x32_f16(false, (A), false, (B), (short)0, (C), false, false)

#if __has_builtin(__builtin_amdgcn_global_load_async_to_lds_b128)
#define HAS_ASYNC_LDS 1
typedef __attribute__((address_space(1))) v4i gas_v4i;   // global (AS1) 16B chunk
typedef __attribute__((address_space(3))) v4i las_v4i;   // LDS    (AS3) 16B chunk
#else
#define HAS_ASYNC_LDS 0
#endif

__device__ __forceinline__ void wait_asynccnt0() {
#if __has_builtin(__builtin_amdgcn_s_wait_asynccnt)
  __builtin_amdgcn_s_wait_asynccnt(0);
#else
  asm volatile("s_wait_asynccnt 0x0" ::: "memory");
#endif
}

// rowPtr = &tile[row][kChunkBase] (16B aligned). kbHi = (lane>=16)?8:0.
// elements 0..7  <-> K = kbHi + 0..7
// elements 8..15 <-> K = kbHi + 16..23   (per CDNA5 16-bit A/B 16x32 layout)
__device__ __forceinline__ v16h load_frag16(const _Float16* rowPtr, int kbHi) {
  Frag16 f;
  f.u[0] = *reinterpret_cast<const v4u*>(rowPtr + kbHi);
  f.u[1] = *reinterpret_cast<const v4u*>(rowPtr + kbHi + 16);
  return f.h;
}

// ---------------- problem constants ----------------
constexpr int    Bn   = 32;
constexpr int    Lseq = 1024;
constexpr int    Dd   = 1024;
constexpr int    Hd   = 256;
constexpr size_t NTOK = (size_t)Bn * Lseq;   // 32768 tokens per tensor
constexpr size_t PREG = NTOK * Dd;           // halves per q/k/v region
constexpr size_t QKVH = 3 * PREG;            // halves per tensor qkv block
constexpr float  SCALE_INV = 1.0f / 32.0f;
constexpr float  LRELU = 0.01f;
constexpr float  BN_EPS = 1e-5f;

// =====================================================================
// Kernel 0: weight transpose+convert to f16, zero output
// W1 [3][1024][256] -> W1hT [3][256][1024] ;  W2 [3][256][1024] -> W2hT [3][1024][256]
// =====================================================================
__global__ __launch_bounds__(256) void prep_kernel(const float* __restrict__ W1,
                                                   const float* __restrict__ W2,
                                                   _Float16* __restrict__ W1hT,
                                                   _Float16* __restrict__ W2hT,
                                                   float* __restrict__ out) {
  size_t i = (size_t)blockIdx.x * 256 + threadIdx.x;
  if (i < (size_t)3 * Dd * Hd) {
    size_t p = i / ((size_t)Dd * Hd), rem = i % ((size_t)Dd * Hd);
    size_t d = rem / Hd, h = rem % Hd;
    W1hT[(p * Hd + h) * Dd + d] = (_Float16)W1[i];
  }
  if (i < (size_t)3 * Hd * Dd) {
    size_t p = i / ((size_t)Hd * Dd), rem = i % ((size_t)Hd * Dd);
    size_t h = rem / Dd, d = rem % Dd;
    W2hT[(p * Dd + d) * Hd + h] = (_Float16)W2[i];
  }
  if (i < (size_t)2 * Bn * Dd) out[i] = 0.0f;
}

// =====================================================================
// Kernel 1: fused 3-net MLP producing q,k,v in f16.
//   q (p=0), k (p=1): [tok][d] row-major
//   v (p=2): transposed [b][d][tok] (for contiguous PV B-fragments)
// block = 256 thr (8 waves), grid = (Ntok/16, 3)
// =====================================================================
__global__ __launch_bounds__(256) void mlp_qkv_kernel(
    const float* __restrict__ x,          // [Ntok][1024] f32
    const _Float16* __restrict__ W1hT,    // [3][256][1024]
    const _Float16* __restrict__ W2hT,    // [3][1024][256]
    const float* __restrict__ b1,         // [3][256]
    const float* __restrict__ g,
    const float* __restrict__ bt,
    const float* __restrict__ rm,
    const float* __restrict__ rv,
    const float* __restrict__ b2,         // [3][1024]
    _Float16* __restrict__ qkv)           // [3*NTOK*Dd] halves (v region transposed)
{
  __shared__ _Float16 Xs[16][Dd];         // 32 KB  input tile (f16)
  __shared__ _Float16 Hs[16][Hd + 8];     // hidden tile (f16), padded stride 264

  const int tid  = threadIdx.x;
  const int lane = tid & 31;
  const int wv   = tid >> 5;
  const int ln15 = lane & 15;
  const int kbHi = (lane >> 4) << 3;      // 0 or 8 (also C-matrix row base)
  const int p    = blockIdx.y;
  const size_t tok0 = (size_t)blockIdx.x * 16;

  // cooperative load + convert X tile (f32 -> f16, must go through VGPRs)
  for (int i = tid; i < 16 * Dd; i += 256) {
    int r = i >> 10, c = i & (Dd - 1);
    Xs[r][c] = (_Float16)x[(tok0 + r) * Dd + c];
  }
  __syncthreads();

  const _Float16* W1p = W1hT + (size_t)p * Hd * Dd;

  // ---- stage 1: H[16,256] = X @ W1 ; each wave does 2 of 16 column-chunks
  for (int cc = 0; cc < 2; ++cc) {
    int ch = wv + cc * 8;                 // h-chunk 0..15
    v8f acc = {0.f, 0.f, 0.f, 0.f, 0.f, 0.f, 0.f, 0.f};
    const _Float16* wrow = W1p + (size_t)(ch * 16 + ln15) * Dd;
    for (int kc = 0; kc < Dd / 32; ++kc) {
      v16h af = load_frag16(&Xs[ln15][kc * 32], kbHi);
      v16h bf = load_frag16(wrow + kc * 32, kbHi);
      acc = WMMA_F16(af, bf, acc);
    }
    int hcol = ch * 16 + ln15;
    float bb = b1[p * Hd + hcol];
    float sc = g[p * Hd + hcol] * rsqrtf(rv[p * Hd + hcol] + BN_EPS);
    float sh = bt[p * Hd + hcol] - rm[p * Hd + hcol] * sc;
    for (int r = 0; r < 8; ++r) {
      float v = acc[r] + bb;
      v = (v > 0.f) ? v : LRELU * v;      // LeakyReLU
      v = v * sc + sh;                    // eval-mode BatchNorm (folded affine)
      Hs[kbHi + r][hcol] = (_Float16)v;
    }
  }
  __syncthreads();

  // ---- stage 2: O[16,1024] = H @ W2 ; each wave owns 8 of 64 column-chunks
  const _Float16* W2p = W2hT + (size_t)p * Dd * Hd;
  _Float16* qk_out = qkv + (size_t)p * PREG;            // for p<2
  _Float16* v_out  = qkv + (size_t)2 * PREG;            // for p==2 (transposed)
  const size_t bi  = tok0 >> 10;
  const size_t ti0 = (tok0 & (Lseq - 1)) + kbHi;

  for (int nc = 0; nc < 8; ++nc) {
    int ch = wv * 8 + nc;                 // d-chunk 0..63
    v8f acc = {0.f, 0.f, 0.f, 0.f, 0.f, 0.f, 0.f, 0.f};
    const _Float16* wrow = W2p + (size_t)(ch * 16 + ln15) * Hd;
    for (int kc = 0; kc < Hd / 32; ++kc) {
      v16h af = load_frag16(&Hs[ln15][kc * 32], kbHi);
      v16h bf = load_frag16(wrow + kc * 32, kbHi);
      acc = WMMA_F16(af, bf, acc);
    }
    int dcol = ch * 16 + ln15;
    float bb = b2[p * Dd + dcol];
    if (p < 2) {
      for (int r = 0; r < 8; ++r)
        qk_out[(tok0 + kbHi + r) * Dd + dcol] = (_Float16)(acc[r] + bb);
    } else {
      Pack8h pk;
      for (int r = 0; r < 8; ++r) pk.t[r] = (_Float16)(acc[r] + bb);
      _Float16* dst = v_out + ((bi * Dd + dcol) * (size_t)Lseq) + ti0;
      *reinterpret_cast<v4u*>(dst) = pk.u;              // 8 contiguous tokens
    }
  }
}

// =====================================================================
// Kernel 2: flash cross-attention. block = 256 thr (8 waves).
// grid = (64 q-tiles, 32 batch, 2 dir). 16 queries/block, 128 keys/iter.
// Q tile staged into LDS via GLOBAL_LOAD_ASYNC_TO_LDS_B128 (ASYNCcnt path).
// Writes partial[qt][dir][bi][d] (deterministic; reduced by kernel 3).
// =====================================================================
__global__ __launch_bounds__(256) void attn_kernel(
    const _Float16* __restrict__ qkvA, const _Float16* __restrict__ qkvB,
    const int* __restrict__ len_a, const int* __restrict__ len_b,
    float* __restrict__ partial)
{
  __shared__ _Float16 Qs[16][Dd];         // 32 KB query tile
  __shared__ float    Ss[16][128];        // score tile
  __shared__ _Float16 Ps[16][128];        // exp(score) tile (f16)
  __shared__ float    mrow[16], lrow[16], arow[16], red[16][8];

  const int tid  = threadIdx.x;
  const int lane = tid & 31;
  const int wv   = tid >> 5;
  const int ln15 = lane & 15;
  const int kbHi = (lane >> 4) << 3;
  const int qt   = blockIdx.x;
  const int bi   = blockIdx.y;
  const int dir  = blockIdx.z;

  const _Float16* Qb = (dir == 0 ? qkvA : qkvB);
  const _Float16* Kt = (dir == 0 ? qkvB : qkvA) + PREG;
  const _Float16* Vt = (dir == 0 ? qkvB : qkvA) + 2 * PREG;
  const int lenK = (dir == 0) ? len_b[bi] : len_a[bi];
  const int lenQ = (dir == 0) ? len_a[bi] : len_b[bi];

  const _Float16* qsrc = Qb + ((size_t)bi * Lseq + (size_t)qt * 16) * Dd;

#if HAS_ASYNC_LDS
  {
    _Float16* lbase = &Qs[0][0];
    // 32 KB tile = 2048 x 16B chunks; 8 async b128 copies per thread, no VGPR data
    for (int i = tid; i < (16 * Dd) / 8; i += 256) {
      __builtin_amdgcn_global_load_async_to_lds_b128(
          (gas_v4i*)(qsrc + (size_t)i * 8),
          (las_v4i*)(lbase + (size_t)i * 8),
          0, 0);
    }
    wait_asynccnt0();
  }
#else
  for (int i = tid; i < 16 * Dd; i += 256) Qs[i >> 10][i & (Dd - 1)] = qsrc[i];
#endif
  if (tid < 16) { mrow[tid] = -1e30f; lrow[tid] = 0.f; }
  __syncthreads();

  v8f o[8];
  for (int j = 0; j < 8; ++j) o[j] = (v8f){0.f, 0.f, 0.f, 0.f, 0.f, 0.f, 0.f, 0.f};

  const _Float16* Kb = Kt + (size_t)bi * Lseq * Dd;
  const _Float16* Vb = Vt + (size_t)bi * Dd * Lseq;

  for (int kt = 0; kt < Lseq / 128; ++kt) {
    // -- S = (Q/32) @ K^T for this wave's 16 keys (full D reduction)
    const int key = kt * 128 + wv * 16 + ln15;    // B-operand column for this lane
    const _Float16* krow = Kb + (size_t)key * Dd;
    if (kt + 1 < Lseq / 128)
      __builtin_prefetch(krow + (size_t)128 * Dd, 0, 1);   // next K tile row
    v8f s = {0.f, 0.f, 0.f, 0.f, 0.f, 0.f, 0.f, 0.f};
    for (int kc = 0; kc < Dd / 32; ++kc) {
      v16h af = load_frag16(&Qs[ln15][kc * 32], kbHi);
      v16h bf = load_frag16(krow + kc * 32, kbHi);
      s = WMMA_F16(af, bf, s);
    }
    const bool kvalid = (key < lenK);
    for (int r = 0; r < 8; ++r)
      Ss[kbHi + r][wv * 16 + ln15] = kvalid ? s[r] * SCALE_INV : -1e9f;
    __syncthreads();

    // -- online softmax stats over the 128-key tile
    if (tid < 128) {
      int rr = tid >> 3, sg = tid & 7;
      float mx = -1e30f;
      for (int j = 0; j < 16; ++j) mx = fmaxf(mx, Ss[rr][sg * 16 + j]);
      red[rr][sg] = mx;
    }
    __syncthreads();
    if (tid < 16) {
      float mn = mrow[tid];
      for (int sg = 0; sg < 8; ++sg) mn = fmaxf(mn, red[tid][sg]);
      arow[tid] = __expf(mrow[tid] - mn);
      mrow[tid] = mn;
    }
    __syncthreads();

    float al[8];
    for (int r = 0; r < 8; ++r) al[r] = arow[kbHi + r];
    for (int j = 0; j < 8; ++j)
      for (int r = 0; r < 8; ++r) o[j][r] *= al[r];

    if (tid < 128) {
      int rr = tid >> 3, sg = tid & 7;
      float mr = mrow[rr], sm = 0.f;
      for (int j = 0; j < 16; ++j) {
        float pv = __expf(Ss[rr][sg * 16 + j] - mr);
        Ps[rr][sg * 16 + j] = (_Float16)pv;
        sm += pv;
      }
      red[rr][sg] = sm;
    }
    __syncthreads();
    if (tid < 16) {
      float sm = 0.f;
      for (int sg = 0; sg < 8; ++sg) sm += red[tid][sg];
      lrow[tid] = lrow[tid] * arow[tid] + sm;
    }
    __syncthreads();

    // -- O += P @ V ; wave owns output columns [wv*128, wv*128+128)
    for (int nc = 0; nc < 8; ++nc) {
      int dcol = wv * 128 + nc * 16 + ln15;       // B-operand column
      const _Float16* vrow = Vb + (size_t)dcol * Lseq + (size_t)kt * 128;
      if (kt + 1 < Lseq / 128)
        __builtin_prefetch(vrow + 128, 0, 1);     // next V segment on this row
      v8f acc = o[nc];
      for (int kc = 0; kc < 4; ++kc) {
        v16h af = load_frag16(&Ps[ln15][kc * 32], kbHi);
        v16h bf = load_frag16(vrow + kc * 32, kbHi);
        acc = WMMA_F16(af, bf, acc);
      }
      o[nc] = acc;
    }
    __syncthreads();
  }

  // finalize: weighted sum over rows = (qmask / (l * lenQ)); pair-reduce lanes
  float wts[8];
  const int q0 = qt * 16;
  for (int r = 0; r < 8; ++r) {
    int qi = q0 + kbHi + r;
    wts[r] = (qi < lenQ) ? 1.0f / (lrow[kbHi + r] * (float)lenQ) : 0.0f;
  }
  float* obase = partial + (((size_t)qt * 2 + dir) * Bn + bi) * Dd;
  for (int nc = 0; nc < 8; ++nc) {
    float sm = 0.f;
    for (int r = 0; r < 8; ++r) sm += o[nc][r] * wts[r];
    sm += __shfl_down(sm, 16, 32);                // rows 8..15 live in lanes 16..31
    if (lane < 16) obase[wv * 128 + nc * 16 + ln15] = sm;
  }
}

// =====================================================================
// Kernel 3: reduce 64 query-tile partials -> out[2][B][D]
// =====================================================================
__global__ __launch_bounds__(256) void reduce_kernel(const float* __restrict__ partial,
                                                     float* __restrict__ out) {
  size_t i = (size_t)blockIdx.x * 256 + threadIdx.x;   // 0..65535
  float s = 0.f;
  for (int qt = 0; qt < 64; ++qt) s += partial[(size_t)qt * (2 * Bn * Dd) + i];
  out[i] = s;
}

// =====================================================================
extern "C" void kernel_launch(void* const* d_in, const int* in_sizes, int n_in,
                              void* d_out, int out_size, void* d_ws, size_t ws_size,
                              hipStream_t stream) {
  const float* a     = (const float*)d_in[0];
  const float* b     = (const float*)d_in[1];
  const float* W1    = (const float*)d_in[2];
  const float* b1    = (const float*)d_in[3];
  const float* g     = (const float*)d_in[4];
  const float* bt    = (const float*)d_in[5];
  const float* rm    = (const float*)d_in[6];
  const float* rv    = (const float*)d_in[7];
  const float* W2    = (const float*)d_in[8];
  const float* b2    = (const float*)d_in[9];
  const int*   len_a = (const int*)d_in[10];
  const int*   len_b = (const int*)d_in[11];
  float* out = (float*)d_out;

  // workspace layout (halves, then float partials)
  _Float16* ws    = (_Float16*)d_ws;
  _Float16* qkvA  = ws;                    // 3*NTOK*Dd halves
  _Float16* qkvB  = ws + QKVH;
  _Float16* W1hT  = ws + 2 * QKVH;         // 3*256*1024
  _Float16* W2hT  = W1hT + (size_t)3 * Hd * Dd;
  float* partials = (float*)(W2hT + (size_t)3 * Dd * Hd);  // 64*2*Bn*Dd f32

  prep_kernel<<<dim3(3072), dim3(256), 0, stream>>>(W1, W2, W1hT, W2hT, out);

  mlp_qkv_kernel<<<dim3(NTOK / 16, 3), dim3(256), 0, stream>>>(
      a, W1hT, W2hT, b1, g, bt, rm, rv, b2, qkvA);
  mlp_qkv_kernel<<<dim3(NTOK / 16, 3), dim3(256), 0, stream>>>(
      b, W1hT, W2hT, b1, g, bt, rm, rv, b2, qkvB);

  attn_kernel<<<dim3(Lseq / 16, Bn, 2), dim3(256), 0, stream>>>(
      qkvA, qkvB, len_a, len_b, partials);

  reduce_kernel<<<dim3((2 * Bn * Dd) / 256), dim3(256), 0, stream>>>(partials, out);
}
